// MultiGCN_28441273434407
// MI455X (gfx1250) — matrix operand order repaired and checked by reading, not compiled, not run on hardware
//
#include <hip/hip_runtime.h>

// ---------------- problem constants (from reference) ----------------
#define N_NODES  100000
#define E_EDGES  400000
#define KSET     2
#define DFEAT    256
#define ATTD     64
#define LLAYERS  3
#define H1DIM    128
#define H2DIM    64
#define EPSV     1e-5f
#define NEGSL    0.01f

// ---------------- WMMA types ----------------
typedef __attribute__((ext_vector_type(16))) __bf16 v16bf;
typedef __attribute__((ext_vector_type(8)))  float  v8f;

union FragBF { unsigned int u[8]; uint4 q[2]; v16bf v; };
union AccF   { float f[8]; v8f v; };

__device__ __forceinline__ unsigned short f2bf(float f) {
  unsigned int u = __float_as_uint(f);
  u += 0x7FFFu + ((u >> 16) & 1u);          // round-to-nearest-even
  return (unsigned short)(u >> 16);
}

// ---------------- elementwise / reduction kernels ----------------
__global__ void zero_f32_kernel(float* p, long long n) {
  long long i = (long long)blockIdx.x * blockDim.x + threadIdx.x;
  long long st = (long long)gridDim.x * blockDim.x;
  for (; i < n; i += st) p[i] = 0.0f;
}

// per-column sum / sumsq accumulation (256 threads = 256 features)
__global__ void bn_reduce_kernel(const float* __restrict__ x, float* __restrict__ sums) {
  int d = threadIdx.x;
  long long r0 = (long long)blockIdx.x * 256;
  float s = 0.f, s2 = 0.f;
  for (int r = 0; r < 256; ++r) {
    long long nn = r0 + r;
    if (nn < N_NODES) {
      float v = x[nn * DFEAT + d];
      s += v; s2 += v * v;
    }
  }
  atomicAdd(&sums[d], s);
  atomicAdd(&sums[DFEAT + d], s2);
}

__global__ void bn_finalize_kernel(const float* __restrict__ sums,
                                   const float* __restrict__ gamma,
                                   const float* __restrict__ beta,
                                   float* __restrict__ ss) {
  int d = threadIdx.x;
  float mu  = sums[d] * (1.0f / N_NODES);
  float var = sums[DFEAT + d] * (1.0f / N_NODES) - mu * mu;
  float inv = rsqrtf(var + EPSV);
  float sc  = gamma[d] * inv;
  ss[d] = sc;
  ss[DFEAT + d] = beta[d] - mu * sc;
}

__global__ void bn_apply_kernel(const float* __restrict__ x,
                                const float* __restrict__ ss,
                                unsigned short* __restrict__ hbf) {
  long long i = (long long)blockIdx.x * blockDim.x + threadIdx.x;
  if (i >= (long long)N_NODES * DFEAT) return;
  int d = (int)(i & (DFEAT - 1));
  hbf[i] = f2bf(x[i] * ss[d] + ss[DFEAT + d]);
}

// deg[k][col] += w  (segment-sum of edge weights)
__global__ void deg_kernel(const int* __restrict__ ei, const float* __restrict__ ew,
                           float* __restrict__ deg) {
  long long i = (long long)blockIdx.x * blockDim.x + threadIdx.x;
  if (i >= (long long)KSET * E_EDGES) return;
  int k = (int)(i / E_EDGES);
  int e = (int)(i % E_EDGES);
  int col = ei[((long long)k * 2 + 1) * E_EDGES + e];
  atomicAdd(&deg[(long long)k * N_NODES + col], ew[i]);
}

__global__ void dinv_kernel(float* __restrict__ deg) {
  long long i = (long long)blockIdx.x * blockDim.x + threadIdx.x;
  if (i >= (long long)KSET * N_NODES) return;
  float v = deg[i];
  deg[i] = (v > 0.0f) ? rsqrtf(v) : 0.0f;
}

// one block per edge; 256 lanes = 256 features; xs[k][col] += norm * xw[row]
__global__ void agg_kernel(const int* __restrict__ ei, const float* __restrict__ ew,
                           const float* __restrict__ dinv, const float* __restrict__ xw,
                           float* __restrict__ xs) {
  __shared__ float s_norm;
  __shared__ int s_row, s_col;
  int e = blockIdx.x;
  int k = blockIdx.y;
  if (threadIdx.x == 0) {
    int row = ei[((long long)k * 2    ) * E_EDGES + e];
    int col = ei[((long long)k * 2 + 1) * E_EDGES + e];
    float w = ew[(long long)k * E_EDGES + e];
    s_row = row; s_col = col;
    s_norm = dinv[(long long)k * N_NODES + col] * w * dinv[(long long)k * N_NODES + row];
  }
  __syncthreads();
  int d = threadIdx.x;
  float v = s_norm * xw[(long long)s_row * DFEAT + d];
  atomicAdd(&xs[((long long)k * N_NODES + s_col) * DFEAT + d], v);
}

// xs += bias ; also emit bf16 copy for the attention GEMM
__global__ void bias_cvt_kernel(float* __restrict__ xs, unsigned short* __restrict__ xsb,
                                const float* __restrict__ b) {
  long long i = (long long)blockIdx.x * blockDim.x + threadIdx.x;
  if (i >= (long long)KSET * N_NODES * DFEAT) return;
  int d = (int)(i & (DFEAT - 1));
  float v = xs[i] + b[d];
  xs[i] = v;
  xsb[i] = f2bf(v);
}

// sim = keys . q  ; alpha = softmax over K=2
__global__ void sim_alpha_kernel(const float* __restrict__ keys, const float* __restrict__ q,
                                 float* __restrict__ alpha) {
  int n = blockIdx.x * blockDim.x + threadIdx.x;
  if (n >= N_NODES) return;
  float s0 = 0.f, s1 = 0.f;
  const float* k0 = keys + (long long)n * ATTD;
  const float* k1 = keys + (long long)N_NODES * ATTD + (long long)n * ATTD;
  for (int h = 0; h < ATTD; ++h) {
    s0 += k0[h] * q[h];
    s1 += k1[h] * q[ATTD + h];
  }
  float m = fmaxf(s0, s1);
  float e0 = expf(s0 - m), e1 = expf(s1 - m);
  float inv = 1.0f / (e0 + e1);
  alpha[n] = e0 * inv;
  alpha[N_NODES + n] = e1 * inv;
}

// h = relu(alpha0*xs0 + alpha1*xs1), emit bf16 for next GEMM
__global__ void combine_kernel(const float* __restrict__ xs, const float* __restrict__ alpha,
                               unsigned short* __restrict__ hbf) {
  long long i = (long long)blockIdx.x * blockDim.x + threadIdx.x;
  if (i >= (long long)N_NODES * DFEAT) return;
  long long n = i >> 8;  // DFEAT=256
  float v = alpha[n] * xs[i] + alpha[N_NODES + n] * xs[(long long)N_NODES * DFEAT + i];
  hbf[i] = f2bf(fmaxf(v, 0.0f));
}

// ---------------- weight packing into B-fragment order ----------------
// B (32x16 bf16) layout: vj, lane l: n = l&15, K = (l>>4)*16 + 2j, 2j+1.
// Packed stream: [kt*nColTiles + ct][lane*8 + j] as u32 (lo|hi<<16) so each
// lane's 8 fragment dwords are contiguous -> two global_load_b128 per fragment.
__global__ void pack_weight_kernel(const float* __restrict__ W, unsigned int* __restrict__ P,
                                   int Kd, int Nc) {
  int idx = blockIdx.x * blockDim.x + threadIdx.x;
  int total = (Kd * Nc) >> 1;
  if (idx >= total) return;
  int j    = idx & 7;
  int lane = (idx >> 3) & 31;
  int tile = idx >> 8;
  int nColTiles = Nc >> 4;
  int ct = tile % nColTiles;
  int kt = tile / nColTiles;
  int n    = lane & 15;
  int koff = (lane >> 4) << 4;
  int k0 = kt * 32 + koff + 2 * j;
  unsigned int lo = f2bf(W[(long long)k0 * Nc + ct * 16 + n]);
  unsigned int hi = f2bf(W[(long long)(k0 + 1) * Nc + ct * 16 + n]);
  P[idx] = lo | (hi << 16);
}

// ---------------- WMMA GEMM: C[MxNc] = A[MxKd] (bf16) * Bpacked ----------------
// block = 256 threads = 8 waves; block tile 128 rows x 64 cols.
// Each wave: one 16-row tile x 4 column tiles => 4 wmma per K-step reusing one
// A fragment (A staged via LDS with coalesced b128; B frags are 2x b128 each).
// EPI: 0 -> f32 raw; 1 -> tanh(acc+bias) f32; 2 -> leaky(acc+bias) bf16; 3 -> leaky(acc+bias) f32
template <int EPI>
__global__ void gemm_wmma_kernel(const unsigned short* __restrict__ A,
                                 const unsigned int* __restrict__ Bp,
                                 const float* __restrict__ bias,
                                 void* __restrict__ Cout,
                                 int M, int Kd, int Nc) {
  __shared__ __align__(16) unsigned short As[128 * 32];  // 8 KB staging tile
  const int wave = threadIdx.x >> 5;
  const int lane = threadIdx.x & 31;
  const int ctBase = blockIdx.x * 4;          // 4 column tiles per block
  const int rowBase = blockIdx.y * 128;
  const int nColTiles = Nc >> 4;
  const int halfsel = lane >> 4;
  const int mloc = lane & 15;

  AccF acc[4];
#pragma unroll
  for (int c = 0; c < 4; ++c)
#pragma unroll
    for (int r = 0; r < 8; ++r) acc[c].f[r] = 0.0f;

  for (int kk = 0; kk < Kd; kk += 32) {
    // stage A: 128 rows x 32 bf16, coalesced b128 loads
#pragma unroll
    for (int p = 0; p < 2; ++p) {
      int r = (threadIdx.x >> 2) + p * 64;
      int c16 = threadIdx.x & 3;
      int grow = rowBase + r;
      if (grow >= M) grow = M - 1;
      const uint4* src = (const uint4*)(A + (long long)grow * Kd + kk + c16 * 8);
      *(uint4*)(&As[r * 32 + c16 * 8]) = *src;
    }
    __syncthreads();

    // A fragment (16x32): lanes 0-15 K in {0..15}, lanes 16-31 K in {16..31}
    FragBF af;
    int am = wave * 16 + mloc;
#pragma unroll
    for (int j = 0; j < 8; ++j) {
      int k0 = (halfsel << 4) + (j << 1);
      af.u[j] = *(const unsigned int*)(&As[am * 32 + k0]);
    }

    // 4 B fragments, each two coalesced b128 loads; 4 wmma reuse the A fragment
#pragma unroll
    for (int c = 0; c < 4; ++c) {
      FragBF bf;
      const uint4* b4 = (const uint4*)(Bp +
          (((long long)(kk >> 5) * nColTiles + (ctBase + c)) << 8) + lane * 8);
      bf.q[0] = b4[0];
      bf.q[1] = b4[1];
      acc[c].v = __builtin_amdgcn_wmma_f32_16x16x32_bf16(
          false, af.v, false, bf.v, (short)0, acc[c].v, false, false);
    }
    __syncthreads();
  }

  // epilogue: C layout -> VGPR r: M=r (lanes 0-15) / M=r+8 (lanes 16-31), N=lane&15
#pragma unroll
  for (int c = 0; c < 4; ++c) {
    int gcol = (ctBase + c) * 16 + mloc;
#pragma unroll
    for (int r = 0; r < 8; ++r) {
      int grow = rowBase + wave * 16 + r + (halfsel << 3);
      if (grow < M) {
        float v = acc[c].f[r];
        if constexpr (EPI != 0) v += bias[gcol];
        if constexpr (EPI == 1) v = tanhf(v);
        if constexpr (EPI == 2 || EPI == 3) v = (v > 0.0f) ? v : NEGSL * v;
        long long oi = (long long)grow * Nc + gcol;
        if constexpr (EPI == 2) ((unsigned short*)Cout)[oi] = f2bf(v);
        else                    ((float*)Cout)[oi] = v;
      }
    }
  }
}

// ---------------- host orchestration ----------------
extern "C" void kernel_launch(void* const* d_in, const int* in_sizes, int n_in,
                              void* d_out, int out_size, void* d_ws, size_t ws_size,
                              hipStream_t stream) {
  const float* x     = (const float*)d_in[0];
  const int*   ei    = (const int*)d_in[1];
  const float* ew    = (const float*)d_in[2];
  const float* gamma = (const float*)d_in[3];
  const float* beta  = (const float*)d_in[4];
  const float* gcnW  = (const float*)d_in[5];
  const float* gcnB  = (const float*)d_in[6];
  const float* attW  = (const float*)d_in[7];
  const float* attB  = (const float*)d_in[8];
  const float* attQ  = (const float*)d_in[9];
  const float* W1    = (const float*)d_in[10];
  const float* b1    = (const float*)d_in[11];
  const float* W2    = (const float*)d_in[12];
  const float* b2    = (const float*)d_in[13];

  char* wsb = (char*)d_ws;
  size_t off = 0;
  auto take = [&](size_t bytes) -> char* {
    char* p = wsb + off;
    off = (off + bytes + 255) & ~(size_t)255;
    return p;
  };
  float* sums = (float*)take(2 * DFEAT * 4);
  float* ss   = (float*)take(2 * DFEAT * 4);
  unsigned short* hbf = (unsigned short*)take((size_t)N_NODES * DFEAT * 2);
  float* xw  = (float*)take((size_t)N_NODES * DFEAT * 4);
  float* deg = (float*)take((size_t)KSET * N_NODES * 4);
  float* xs  = (float*)take((size_t)KSET * N_NODES * DFEAT * 4);
  unsigned short* xsb = (unsigned short*)take((size_t)KSET * N_NODES * DFEAT * 2);
  float* keys  = (float*)take((size_t)KSET * N_NODES * ATTD * 4);
  float* alpha = (float*)take((size_t)KSET * N_NODES * 4);
  unsigned short* t1 = (unsigned short*)take((size_t)N_NODES * H1DIM * 2);
  unsigned int* gcnP = (unsigned int*)take((size_t)LLAYERS * DFEAT * DFEAT * 2);
  unsigned int* attP = (unsigned int*)take((size_t)LLAYERS * KSET * DFEAT * ATTD * 2);
  unsigned int* p1P  = (unsigned int*)take((size_t)DFEAT * H1DIM * 2);
  unsigned int* p2P  = (unsigned int*)take((size_t)H1DIM * H2DIM * 2);

  auto packW = [&](const float* W, unsigned int* P, int Kd, int Nc) {
    int total = (Kd * Nc) / 2;
    pack_weight_kernel<<<(total + 255) / 256, 256, 0, stream>>>(W, P, Kd, Nc);
  };

  // --- batchnorm ---
  zero_f32_kernel<<<4, 256, 0, stream>>>(sums, 2 * DFEAT);
  bn_reduce_kernel<<<(N_NODES + 255) / 256, 256, 0, stream>>>(x, sums);
  bn_finalize_kernel<<<1, 256, 0, stream>>>(sums, gamma, beta, ss);
  bn_apply_kernel<<<(int)(((long long)N_NODES * DFEAT + 255) / 256), 256, 0, stream>>>(x, ss, hbf);

  // --- pack all weights (bf16, B-fragment order) ---
  for (int i = 0; i < LLAYERS; ++i)
    packW(gcnW + (size_t)i * DFEAT * DFEAT, gcnP + (size_t)i * DFEAT * DFEAT / 2, DFEAT, DFEAT);
  for (int i = 0; i < LLAYERS; ++i)
    for (int k = 0; k < KSET; ++k)
      packW(attW + ((size_t)i * KSET + k) * DFEAT * ATTD,
            attP + ((size_t)i * KSET + k) * DFEAT * ATTD / 2, DFEAT, ATTD);
  packW(W1, p1P, DFEAT, H1DIM);
  packW(W2, p2P, H1DIM, H2DIM);

  // --- degree / normalization (edge-structure invariant across layers) ---
  zero_f32_kernel<<<512, 256, 0, stream>>>(deg, (long long)KSET * N_NODES);
  deg_kernel<<<(int)(((long long)KSET * E_EDGES + 255) / 256), 256, 0, stream>>>(ei, ew, deg);
  dinv_kernel<<<(int)(((long long)KSET * N_NODES + 255) / 256), 256, 0, stream>>>(deg);

  const int rowBlocks = (N_NODES + 127) / 128;
  for (int i = 0; i < LLAYERS; ++i) {
    // xw = h @ gcn_W[i]
    gemm_wmma_kernel<0><<<dim3(DFEAT / 64, rowBlocks), 256, 0, stream>>>(
        hbf, gcnP + (size_t)i * DFEAT * DFEAT / 2, nullptr, xw, N_NODES, DFEAT, DFEAT);
    // xs = scatter-aggregate
    zero_f32_kernel<<<8192, 256, 0, stream>>>(xs, (long long)KSET * N_NODES * DFEAT);
    agg_kernel<<<dim3(E_EDGES, KSET), 256, 0, stream>>>(ei, ew, deg, xw, xs);
    // xs += b ; bf16 copy
    bias_cvt_kernel<<<(int)(((long long)KSET * N_NODES * DFEAT + 255) / 256), 256, 0, stream>>>(
        xs, xsb, gcnB + (size_t)i * DFEAT);
    // keys = tanh(xs @ att_W + att_b)
    for (int k = 0; k < KSET; ++k)
      gemm_wmma_kernel<1><<<dim3(ATTD / 64, rowBlocks), 256, 0, stream>>>(
          xsb + (size_t)k * N_NODES * DFEAT,
          attP + ((size_t)i * KSET + k) * DFEAT * ATTD / 2,
          attB + ((size_t)i * KSET + k) * ATTD,
          keys + (size_t)k * N_NODES * ATTD, N_NODES, DFEAT, ATTD);
    // alpha = softmax_k(keys . q)
    sim_alpha_kernel<<<(N_NODES + 255) / 256, 256, 0, stream>>>(
        keys, attQ + (size_t)i * KSET * ATTD, alpha);
    // h = relu(sum_k alpha_k * xs_k)
    combine_kernel<<<(int)(((long long)N_NODES * DFEAT + 255) / 256), 256, 0, stream>>>(
        xs, alpha, hbf);
  }

  // --- projection head ---
  gemm_wmma_kernel<2><<<dim3(H1DIM / 64, rowBlocks), 256, 0, stream>>>(
      hbf, p1P, b1, t1, N_NODES, DFEAT, H1DIM);
  gemm_wmma_kernel<3><<<dim3(H2DIM / 64, rowBlocks), 256, 0, stream>>>(
      t1, p2P, b2, d_out, N_NODES, H1DIM, H2DIM);
}
